// ATTNLayer_57638461112468
// MI455X (gfx1250) — compile-verified, matrix operand order checked by reading, not run on hardware
//
#include <hip/hip_runtime.h>
#include <hip/hip_bf16.h>
#include <math.h>

// ---------------------------------------------------------------------------
// CDNA5 (gfx1250) attention layer, fp16 WMMA path (wave32).
// Core: v_wmma_f32_16x16x32_f16 GEMM, fragment-ready LDS layout, 64x64 wave
// tiles (16 WMMA per wave per K-step at 1.0 ds_load_b128 per WMMA).
// ---------------------------------------------------------------------------

typedef __attribute__((ext_vector_type(16))) _Float16 v16h;
typedef __attribute__((ext_vector_type(8)))  _Float16 v8h;
typedef __attribute__((ext_vector_type(8)))  float    v8f;

#define EPSF 1e-6f
#define BASE_THETA 10000.0f

// ---------------- RMS norm: x (fp32, S x E) -> xq (fp16) --------------------
__global__ __launch_bounds__(256)
void rmsnorm_kernel(const float* __restrict__ x, _Float16* __restrict__ xq, int E) {
    const int row = blockIdx.x;
    const size_t base = (size_t)row * E;
    __shared__ float red[8];

    float ss = 0.0f;
    for (int c = threadIdx.x; c < E; c += 256) {
        float v = x[base + c];
        ss += v * v;
    }
    #pragma unroll
    for (int off = 16; off > 0; off >>= 1) ss += __shfl_xor(ss, off, 32);
    if ((threadIdx.x & 31) == 0) red[threadIdx.x >> 5] = ss;
    __syncthreads();
    float tot = 0.0f;
    #pragma unroll
    for (int i = 0; i < 8; ++i) tot += red[i];
    const float inv = rsqrtf(tot / (float)E + EPSF);
    for (int c = threadIdx.x; c < E; c += 256)
        xq[base + c] = (_Float16)(x[base + c] * inv);
}

// ------------- weight quantize: w16[r,c] = fp16(scale[r] * w[r,c]) ----------
__global__ __launch_bounds__(256)
void cvt_w_kernel(const float* __restrict__ w, const float* __restrict__ scale,
                  _Float16* __restrict__ out, int E, int use_scale) {
    size_t i = (size_t)blockIdx.x * 256 + threadIdx.x;
    size_t n = (size_t)E * E;
    if (i >= n) return;
    int r = (int)(i / (size_t)E);
    float s = use_scale ? scale[r] : 1.0f;
    out[i] = (_Float16)(s * w[i]);
}

// ---------------- RoPE applied in place to q and k (fp16) -------------------
__global__ __launch_bounds__(256)
void rope_kernel(_Float16* __restrict__ q, _Float16* __restrict__ k,
                 int S, int E) {
    size_t idx = (size_t)blockIdx.x * 256 + threadIdx.x;   // over S * E/2 pairs
    const int halfE = E >> 1;
    if (idx >= (size_t)S * halfE) return;
    int row = (int)(idx / halfE);
    int j   = (int)(idx % halfE);
    float freq  = __expf(-(2.0f * (float)j / (float)E) * __logf(BASE_THETA));
    float theta = (float)row * freq;
    float c, s;
    __sincosf(theta, &s, &c);
    size_t base = (size_t)row * E + 2 * (size_t)j;

    float qe = (float)q[base], qo = (float)q[base + 1];
    q[base]     = (_Float16)( qe * c + qo * s);
    q[base + 1] = (_Float16)(-qe * s + qo * c);

    float ke = (float)k[base], ko = (float)k[base + 1];
    k[base]     = (_Float16)( ke * c + ko * s);
    k[base + 1] = (_Float16)(-ke * s + ko * c);
}

// ---------------- causal softmax over fp16 scores (S x S), fp32 math --------
__global__ __launch_bounds__(256)
void softmax_causal_kernel(_Float16* __restrict__ sc, int S) {
    const int row = blockIdx.x;
    const size_t base = (size_t)row * S;
    __shared__ float red[8];

    float m = -INFINITY;
    for (int c = threadIdx.x; c < S; c += 256) {
        float v = (c <= row) ? (float)sc[base + c] : -INFINITY;
        m = fmaxf(m, v);
    }
    #pragma unroll
    for (int off = 16; off > 0; off >>= 1) m = fmaxf(m, __shfl_xor(m, off, 32));
    if ((threadIdx.x & 31) == 0) red[threadIdx.x >> 5] = m;
    __syncthreads();
    float bm = red[0];
    #pragma unroll
    for (int i = 1; i < 8; ++i) bm = fmaxf(bm, red[i]);
    __syncthreads();

    float sum = 0.0f;
    for (int c = threadIdx.x; c < S; c += 256) {
        float v = (c <= row) ? __expf((float)sc[base + c] - bm) : 0.0f;
        sc[base + c] = (_Float16)v;
        sum += v;
    }
    #pragma unroll
    for (int off = 16; off > 0; off >>= 1) sum += __shfl_xor(sum, off, 32);
    if ((threadIdx.x & 31) == 0) red[threadIdx.x >> 5] = sum;
    __syncthreads();
    float tot = 0.0f;
    #pragma unroll
    for (int i = 0; i < 8; ++i) tot += red[i];
    const float inv = 1.0f / tot;

    for (int c = threadIdx.x; c < S; c += 256)
        sc[base + c] = (_Float16)((float)sc[base + c] * inv);
}

// ---------------------------------------------------------------------------
// WMMA GEMM:  C(MxN) = alpha * A(MxK,f16,row-major) * B
//   BT=false : B is K x N row-major (NN)
//   BT=true  : B is N x K row-major (NT, i.e. C = A * B^T)
//
// LDS tiles in fragment-ready layout:
//   - row-major per M (A) / per N (B), padded stride LDK = BK + 8 halfs
//     (80 B: keeps 16 B alignment for b128, staggers banks)
//   - within a row, K blocks [8..16) and [16..24) are swapped so the CDNA5
//     16-bit fragment for lane-half h (K = {h*8.., 16+h*8..}) is the
//     CONTIGUOUS 32-byte run tile[row*LDK + h*16 .. +16)
//     -> two ds_load_b128 straight into WMMA source registers.
//
// Block tile 128x256, K-step 32, 256 threads = 8 waves in a 2(M) x 4(N) grid;
// each wave computes 64x64 via 4x4 v_wmma_f32_16x16x32_f16 accumulators
// (16 WMMA per K-step, 16 ds_load_b128 per K-step per lane).
// ---------------------------------------------------------------------------
constexpr int BM = 128, BN = 256, BK = 32;
constexpr int LDK = BK + 8;   // padded halfs per LDS row

// permuted position of K within an LDS row (swap middle 8-blocks)
__device__ __forceinline__ int kpos(int k) {
    int blk = k >> 3;                                     // 0..3
    int swapped = blk ^ ((blk == 1 || blk == 2) ? 3 : 0); // 0,2,1,3
    return (swapped << 3) | (k & 7);
}

template <typename CT, bool BT>
__global__ __launch_bounds__(256)
void gemm_f16_wmma(const _Float16* __restrict__ A, const _Float16* __restrict__ B,
                   CT* __restrict__ C, int M, int N, int K, float alpha) {
    __shared__ _Float16 As[BM * LDK];   // m-major, permuted-K (10 KB)
    __shared__ _Float16 Bs[BN * LDK];   // n-major, permuted-K (20 KB)

    const int tid  = threadIdx.x;
    const int wave = tid >> 5;
    const int lane = tid & 31;
    const int waveM = wave >> 2;         // 0..1  (64 rows each)
    const int waveN = wave & 3;          // 0..3  (64 cols each)
    const int half  = lane >> 4;         // 0..1
    const int mrow  = lane & 15;         // 0..15

    const int blockRow = blockIdx.y * BM;
    const int blockCol = blockIdx.x * BN;

    v8f acc[4][4] = {};

    for (int k0 = 0; k0 < K; k0 += BK) {
        // ---- stage A tile: 128x32 halfs; thread covers 16 halfs of one row ----
        {
            int r = tid >> 1, seg = tid & 1;            // seg: K halves 0..15 / 16..31
            const _Float16* gp = A + (size_t)(blockRow + r) * K + k0 + seg * 16;
            // permuted dst segments: {0,1}->{0,2}, {2,3}->{1,3} (8-half units)
            int d0 = seg ? 8 : 0, d1 = seg ? 24 : 16;
            *(v8h*)(&As[r * LDK + d0]) = *(const v8h*)(gp);
            *(v8h*)(&As[r * LDK + d1]) = *(const v8h*)(gp + 8);
            if (k0 + BK < K) __builtin_prefetch(gp + BK, 0, 3);
        }
        // ---- stage B tile: 256 n-rows x 32 K halfs ----
        if (BT) {
            int n = tid;                                 // one full n-row per thread
            const _Float16* gp = B + (size_t)(blockCol + n) * K + k0;
            _Float16* dp = &Bs[n * LDK];
            // src 8-half segs {0,1,2,3} -> dst offsets {0,16,8,24}
            *(v8h*)(dp + 0)  = *(const v8h*)(gp + 0);
            *(v8h*)(dp + 16) = *(const v8h*)(gp + 8);
            *(v8h*)(dp + 8)  = *(const v8h*)(gp + 16);
            *(v8h*)(dp + 24) = *(const v8h*)(gp + 24);
            if (k0 + BK < K) __builtin_prefetch(gp + BK, 0, 3);
        } else {
            // B row-major K x N: coalesced row read, scalar transpose store
            int kk = tid >> 3, n0 = (tid & 7) * 32;      // 32 rows x 8 segs of 32
            const _Float16* gp = B + (size_t)(k0 + kk) * N + blockCol + n0;
            int p = kpos(kk);
            #pragma unroll
            for (int sgi = 0; sgi < 4; ++sgi) {
                v8h vseg = *(const v8h*)(gp + sgi * 8);
                #pragma unroll
                for (int i = 0; i < 8; ++i)
                    Bs[(n0 + sgi * 8 + i) * LDK + p] = vseg[i];
            }
            if (k0 + BK < K) __builtin_prefetch(gp + (size_t)BK * N, 0, 3);
        }
        __syncthreads();

        // ---- fragments: one contiguous 32B run per lane -> 2x ds_load_b128 ----
        v16h afrag[4], bfrag[4];
        #pragma unroll
        for (int mi = 0; mi < 4; ++mi) {
            int r = waveM * 64 + mi * 16 + mrow;
            afrag[mi] = *(const v16h*)(&As[r * LDK + half * 16]);
        }
        #pragma unroll
        for (int ni = 0; ni < 4; ++ni) {
            int n = waveN * 64 + ni * 16 + mrow;
            bfrag[ni] = *(const v16h*)(&Bs[n * LDK + half * 16]);
        }
        #pragma unroll
        for (int mi = 0; mi < 4; ++mi)
            #pragma unroll
            for (int ni = 0; ni < 4; ++ni)
                acc[mi][ni] = __builtin_amdgcn_wmma_f32_16x16x32_f16(
                    /*neg_a=*/false, afrag[mi], /*neg_b=*/false, bfrag[ni],
                    /*c_mod=*/(short)0, acc[mi][ni],
                    /*reuse_a=*/false, /*reuse_b=*/false);
        __syncthreads();
    }

    // ---- store: C/D layout — lane: N=mrow, VGPR r: M = r + half*8 ----
    #pragma unroll
    for (int mi = 0; mi < 4; ++mi) {
        int rbase = blockRow + waveM * 64 + mi * 16 + half * 8;
        #pragma unroll
        for (int ni = 0; ni < 4; ++ni) {
            int col = blockCol + waveN * 64 + ni * 16 + mrow;
            #pragma unroll
            for (int r = 0; r < 8; ++r) {
                float v = acc[mi][ni][r] * alpha;
                C[(size_t)(rbase + r) * N + col] = (CT)(_Float16)v;  // round to f16 first
            }
        }
    }
}

// ---------------------------------------------------------------------------
extern "C" void kernel_launch(void* const* d_in, const int* in_sizes, int n_in,
                              void* d_out, int out_size, void* d_ws, size_t ws_size,
                              hipStream_t stream) {
    (void)n_in; (void)out_size; (void)ws_size;

    const float* x     = (const float*)d_in[0];
    const float* w_q   = (const float*)d_in[1];
    const float* w_k   = (const float*)d_in[2];
    const float* w_v   = (const float*)d_in[3];
    const float* w_out = (const float*)d_in[4];
    const float* scal  = (const float*)d_in[5];

    const int E = in_sizes[5];                 // 4096
    const int S = in_sizes[0] / E;             // 4096

    // workspace layout (fp16 buffers, 256B-aligned); peak 8 x S*E halfs
    char* ws = (char*)d_ws;
    const size_t act = ((size_t)S * E * sizeof(_Float16) + 255) & ~(size_t)255;
    _Float16* xq   = (_Float16*)(ws + 0 * act);
    _Float16* wq16 = (_Float16*)(ws + 1 * act);
    _Float16* wk16 = (_Float16*)(ws + 2 * act);
    _Float16* wv16 = (_Float16*)(ws + 3 * act);
    _Float16* wo16 = (_Float16*)(ws + 4 * act);
    _Float16* q    = (_Float16*)(ws + 5 * act);
    _Float16* k    = (_Float16*)(ws + 6 * act);
    _Float16* v    = (_Float16*)(ws + 7 * act);
    _Float16* sc   = wq16;   // reuse: weights wq/wk dead after projections
    _Float16* attn = wk16;

    // 1. RMS norm -> fp16
    rmsnorm_kernel<<<S, 256, 0, stream>>>(x, xq, E);

    // 2. weight row-scale + quantize
    {
        size_t n = (size_t)E * E;
        int blocks = (int)((n + 255) / 256);
        cvt_w_kernel<<<blocks, 256, 0, stream>>>(w_q,   scal, wq16, E, 1);
        cvt_w_kernel<<<blocks, 256, 0, stream>>>(w_k,   scal, wk16, E, 1);
        cvt_w_kernel<<<blocks, 256, 0, stream>>>(w_v,   scal, wv16, E, 1);
        cvt_w_kernel<<<blocks, 256, 0, stream>>>(w_out, scal, wo16, E, 0);
    }

    // 3. projections: q/k/v = xq @ w*  (NN, fp16 out)
    dim3 gProj(E / BN, S / BM);
    gemm_f16_wmma<_Float16, false><<<gProj, 256, 0, stream>>>(xq, wq16, q, S, E, E, 1.0f);
    gemm_f16_wmma<_Float16, false><<<gProj, 256, 0, stream>>>(xq, wk16, k, S, E, E, 1.0f);
    gemm_f16_wmma<_Float16, false><<<gProj, 256, 0, stream>>>(xq, wv16, v, S, E, E, 1.0f);

    // 4. RoPE on q, k
    {
        size_t pairs = (size_t)S * (E / 2);
        rope_kernel<<<(int)((pairs + 255) / 256), 256, 0, stream>>>(q, k, S, E);
    }

    // 5. scores = (q @ k^T) / sqrt(E)  (NT), causal softmax
    dim3 gScore(S / BN, S / BM);
    gemm_f16_wmma<_Float16, true><<<gScore, 256, 0, stream>>>(
        q, k, sc, S, S, E, 1.0f / sqrtf((float)E));
    softmax_causal_kernel<<<S, 256, 0, stream>>>(sc, S);

    // 6. attn = softmax(scores) @ v  (NN)
    gemm_f16_wmma<_Float16, false><<<gProj, 256, 0, stream>>>(sc, v, attn, S, E, S, 1.0f);

    // 7. out = attn @ wo (NN), fp32 store (value rounded to fp16 first)
    gemm_f16_wmma<float, false><<<gProj, 256, 0, stream>>>(
        attn, wo16, (float*)d_out, S, E, E, 1.0f);
}
